// Focus_78417512890912
// MI455X (gfx1250) — matrix operand order repaired and checked by reading, not compiled, not run on hardware
//
#include <hip/hip_runtime.h>
#include <hip/hip_bf16.h>
#include <math.h>

// ---------------------------------------------------------------------------
// MI455X (gfx1250) implementation.
//   b=32, t=2048, D=100 (pad 128), H=2, K=50 (pad 64).
//   Needs ~84.3 MB of d_ws scratch.
// ---------------------------------------------------------------------------

typedef _Float16 half_t;
typedef __attribute__((ext_vector_type(16))) _Float16 v16h;
typedef __attribute__((ext_vector_type(8)))  _Float16 v8h;
typedef __attribute__((ext_vector_type(8)))  float    v8f;

#define BB    32
#define TT    2048
#define BT    65536          // BB*TT
#define DPAD  128            // D=100 padded to 4 k-chunks of 32
#define NPAD  112            // 100 out-features padded to 7 tiles of 16
#define HD    64             // head dim 50 padded
#define INV_SCALE 0.316227766016838f   // 1 / 100**0.25

union V16 { v16h v; v8h h[2]; };

// load 16 halves as two 8-half chunks, second at +off1 halves
static __device__ __forceinline__ v16h ld16_2(const half_t* p, int off1) {
    V16 u;
    u.h[0] = *(const v8h*)p;
    u.h[1] = *(const v8h*)(p + off1);
    return u.v;
}

#define WMMA(A, B, C) __builtin_amdgcn_wmma_f32_16x16x32_f16( \
        false, (A), false, (B), (short)0, (C), false, false)

// pe quirk: pe[:b] -> position index == batch index, broadcast over t
static __device__ __forceinline__ float pe_val(int pos, int c) {
    int pair = c >> 1;
    float dv = __expf(-9.210340371976184f * (float)pair / 50.0f); // 10000^(-2i/100)
    float a  = (float)pos * dv;
    return (c & 1) ? __cosf(a) : __sinf(a);
}

// ---------------------------------------------------------------------------
__global__ void k_zero(uint4* p, int n) {
    int i = blockIdx.x * blockDim.x + threadIdx.x;
    if (i < n) p[i] = make_uint4(0u, 0u, 0u, 0u);
}

// pack Wq,Wk,Wv,Wu ([100][100] row-major = [out][in]) -> f16 [4][112][128],
// folding 1/SCALE into Wq and Wk
__global__ void k_prep_w(const float* Wq, const float* Wk, const float* Wv,
                         const float* Wu, half_t* wpack) {
    int i = blockIdx.x * blockDim.x + threadIdx.x;
    if (i >= 4 * NPAD * DPAD) return;
    int w = i / (NPAD * DPAD);
    int r = (i / DPAD) % NPAD;
    int c = i & (DPAD - 1);
    float v = 0.f;
    if (r < 100 && c < 100) {
        const float* W = (w == 0) ? Wq : (w == 1) ? Wk : (w == 2) ? Wv : Wu;
        v = W[r * 100 + c];
        if (w < 2) v *= INV_SCALE;
    }
    wpack[i] = (half_t)v;
}

// h = emb[x] + pe(batch)  -> f16 [BT][128] (zero padded)
__global__ void k_build_h(const int* x, const float* emb, half_t* h16) {
    int i  = blockIdx.x * blockDim.x + threadIdx.x;   // BT*128 threads
    int bt = i >> 7, c = i & 127;
    float v = 0.f;
    if (c < 100) v = emb[x[bt] * 100 + c] + pe_val(bt >> 11, c);
    h16[i] = (half_t)v;
}

// q/k/v projection: h16 [BT][128] @ W[out][in] -> q/k [bh][t][64], v^T [bh][64][t]
__global__ void __launch_bounds__(128) k_qkv(const half_t* __restrict__ h16,
                                             const half_t* __restrict__ wpack,
                                             half_t* qb, half_t* kb, half_t* vtb) {
    __shared__ __align__(16) half_t wl[NPAD * DPAD];
    const int which = blockIdx.y;                 // 0=q 1=k 2=v
    const half_t* wsrc = wpack + (size_t)which * NPAD * DPAD;
    for (int i = threadIdx.x; i < NPAD * DPAD / 8; i += 128)
        ((v8h*)wl)[i] = ((const v8h*)wsrc)[i];
    __syncthreads();

    int lane = threadIdx.x & 31, w = threadIdx.x >> 5;
    int idx = blockIdx.x * 4 + w;                 // 4096*7 waves exactly
    int mt = idx / 7, nt = idx % 7;
    int ln = lane & 15, lh = lane >> 4;

    const half_t* arow = h16 + (size_t)(mt * 16 + ln) * DPAD + lh * 8;
    const half_t* brow = wl + (nt * 16 + ln) * DPAD + lh * 8;

    v8f acc = {};
    acc = WMMA(ld16_2(arow +  0, 16), ld16_2(brow +  0, 16), acc);
    acc = WMMA(ld16_2(arow + 32, 16), ld16_2(brow + 32, 16), acc);
    acc = WMMA(ld16_2(arow + 64, 16), ld16_2(brow + 64, 16), acc);
    acc = WMMA(ld16_2(arow + 96, 16), ld16_2(brow + 96, 16), acc);

    int col = nt * 16 + ln;                       // D-frag: N fixed per lane
    if (col < 100) {
        int head = col >= 50;
        int kk = col - head * 50;
        #pragma unroll
        for (int i = 0; i < 8; i++) {             // M = i + lh*8
            int bt = mt * 16 + i + lh * 8;
            int b = bt >> 11, tok = bt & 2047;
            int bh = b * 2 + head;
            half_t hv = (half_t)acc[i];
            if (which == 2)      vtb[((size_t)bh * HD + kk) * TT + tok] = hv;
            else if (which == 1) kb [((size_t)bh * TT + tok) * HD + kk] = hv;
            else                 qb [((size_t)bh * TT + tok) * HD + kk] = hv;
        }
    }
}

// flash attention: one wave per (bh, 16-query tile); key blocks of 32.
// Per-wave private LDS tile for the P bounce: LDS ops from one wave complete
// in order (DScnt is in-order per wave), so no block barrier is needed --
// only a compile-time scheduling fence to keep the store->load order.
__global__ void __launch_bounds__(128) k_attn(const half_t* __restrict__ qb,
                                              const half_t* __restrict__ kb,
                                              const half_t* __restrict__ vtb,
                                              const float*  __restrict__ mask,
                                              half_t* __restrict__ ctxh) {
    __shared__ __align__(16) half_t plds[4][16 * 32];
    int lane = threadIdx.x & 31, w = threadIdx.x >> 5;
    int gw = blockIdx.x * 4 + w;                  // 64*128 waves exactly
    int bh = gw >> 7, qt = gw & 127;
    int ln = lane & 15, lh = lane >> 4;

    const half_t* qp = qb  + (size_t)bh * TT * HD;
    const half_t* kp = kb  + (size_t)bh * TT * HD;
    const half_t* vp = vtb + (size_t)bh * HD * TT;
    const float* mrow = mask + (size_t)(bh & 31) * TT;  // concat([mask,mask])[bh]
    half_t* pl = plds[w];

    const half_t* aqr = qp + (size_t)(qt * 16 + ln) * HD + lh * 8;
    v16h a0 = ld16_2(aqr,      16);
    v16h a1 = ld16_2(aqr + 32, 16);

    float m_run[8], l_run[8];
    v8f acc[4] = {{}, {}, {}, {}};
    #pragma unroll
    for (int i = 0; i < 8; i++) { m_run[i] = -1e30f; l_run[i] = 0.f; }

    for (int KB = 0; KB < TT; KB += 32) {
        if (KB + 32 < TT) {                       // -> global_prefetch_b8
            __builtin_prefetch(kp + (size_t)(KB + 32 + ln) * HD, 0, 1);
            __builtin_prefetch(vp + (size_t)ln * TT + KB + 32, 0, 1);
        }
        // S tiles: Q(16x64) @ K^T -> two 16x16 f32 tiles (32 keys)
        const half_t* kr0 = kp + (size_t)(KB + ln)      * HD + lh * 8;
        const half_t* kr1 = kp + (size_t)(KB + 16 + ln) * HD + lh * 8;
        v8f s0 = {}, s1 = {};
        s0 = WMMA(a0, ld16_2(kr0,      16), s0);
        s0 = WMMA(a1, ld16_2(kr0 + 32, 16), s0);
        s1 = WMMA(a0, ld16_2(kr1,      16), s1);
        s1 = WMMA(a1, ld16_2(kr1 + 32, 16), s1);

        float mv0 = mrow[KB + ln], mv1 = mrow[KB + 16 + ln];
        #pragma unroll
        for (int i = 0; i < 8; i++) {             // per-row flash update
            float e0 = s0[i] + mv0, e1 = s1[i] + mv1;
            float bm = fmaxf(e0, e1);
            bm = fmaxf(bm, __shfl_xor(bm, 1, 32));
            bm = fmaxf(bm, __shfl_xor(bm, 2, 32));
            bm = fmaxf(bm, __shfl_xor(bm, 4, 32));
            bm = fmaxf(bm, __shfl_xor(bm, 8, 32));
            float mn   = fmaxf(m_run[i], bm);
            float corr = __expf(m_run[i] - mn);
            float p0 = __expf(e0 - mn), p1 = __expf(e1 - mn);
            float rs = p0 + p1;
            rs += __shfl_xor(rs, 1, 32);
            rs += __shfl_xor(rs, 2, 32);
            rs += __shfl_xor(rs, 4, 32);
            rs += __shfl_xor(rs, 8, 32);
            l_run[i] = l_run[i] * corr + rs;
            m_run[i] = mn;
            acc[0][i] *= corr; acc[1][i] *= corr;
            acc[2][i] *= corr; acc[3][i] *= corr;
            int row = i + lh * 8;                 // bounce P to LDS for A-frag
            pl[row * 32 + ln]      = (half_t)p0;
            pl[row * 32 + 16 + ln] = (half_t)p1;
        }
        __builtin_amdgcn_wave_barrier();          // keep DS store -> load order
        v16h ap = ld16_2(pl + ln * 32 + lh * 8, 16);   // P A-frag 16x32
        __builtin_amdgcn_wave_barrier();
        const half_t* vr = vp + KB + lh * 8;
        acc[0] = WMMA(ap, ld16_2(vr + (size_t)(ln)      * TT, 16), acc[0]);
        acc[1] = WMMA(ap, ld16_2(vr + (size_t)(16 + ln) * TT, 16), acc[1]);
        acc[2] = WMMA(ap, ld16_2(vr + (size_t)(32 + ln) * TT, 16), acc[2]);
        acc[3] = WMMA(ap, ld16_2(vr + (size_t)(48 + ln) * TT, 16), acc[3]);
    }

    int b = bh >> 1, head = bh & 1;
    #pragma unroll
    for (int dt = 0; dt < 4; dt++) {
        int dim = dt * 16 + ln;
        if (dim < 50) {
            #pragma unroll
            for (int i = 0; i < 8; i++) {
                int tok = qt * 16 + i + lh * 8;
                ctxh[(size_t)(b * TT + tok) * DPAD + head * 50 + dim] =
                    (half_t)(acc[dt][i] / l_run[i]);
            }
        }
    }
}

// x1 = h + ctx @ Wu^T + bu  (h recomputed from emb+pe), written straight to d_out
__global__ void __launch_bounds__(128) k_wu(const half_t* __restrict__ ctxh,
                                            const half_t* __restrict__ wpack,
                                            const float* bu, const int* x,
                                            const float* emb, float* x1) {
    __shared__ __align__(16) half_t wl[NPAD * DPAD];
    const half_t* wsrc = wpack + (size_t)3 * NPAD * DPAD;
    for (int i = threadIdx.x; i < NPAD * DPAD / 8; i += 128)
        ((v8h*)wl)[i] = ((const v8h*)wsrc)[i];
    __syncthreads();

    int lane = threadIdx.x & 31, w = threadIdx.x >> 5;
    int idx = blockIdx.x * 4 + w;
    int mt = idx / 7, nt = idx % 7;
    int ln = lane & 15, lh = lane >> 4;

    const half_t* arow = ctxh + (size_t)(mt * 16 + ln) * DPAD + lh * 8;
    const half_t* brow = wl + (nt * 16 + ln) * DPAD + lh * 8;
    v8f acc = {};
    acc = WMMA(ld16_2(arow +  0, 16), ld16_2(brow +  0, 16), acc);
    acc = WMMA(ld16_2(arow + 32, 16), ld16_2(brow + 32, 16), acc);
    acc = WMMA(ld16_2(arow + 64, 16), ld16_2(brow + 64, 16), acc);
    acc = WMMA(ld16_2(arow + 96, 16), ld16_2(brow + 96, 16), acc);

    int col = nt * 16 + ln;
    if (col < 100) {
        float bucol = bu[col];
        #pragma unroll
        for (int i = 0; i < 8; i++) {
            int bt = mt * 16 + i + lh * 8;
            float hv = emb[x[bt] * 100 + col] + pe_val(bt >> 11, col);
            x1[(size_t)bt * 100 + col] = acc[i] + bucol + hv;
        }
    }
}

// logits = relu(x1@W1^T+b1)@W2^T + b2 + mask     (one thread per token)
__global__ void k_mlp(const float* __restrict__ x1, const float* __restrict__ W1,
                      const float* __restrict__ b1, const float* __restrict__ W2,
                      const float* __restrict__ b2, const float* __restrict__ mask,
                      float* __restrict__ logits) {
    int bt = blockIdx.x * blockDim.x + threadIdx.x;
    const float* xp = x1 + (size_t)bt * 100;
    float xr[100];
    #pragma unroll 4
    for (int c = 0; c < 100; c++) xr[c] = xp[c];
    float y = b2[0];
    for (int j = 0; j < 50; j++) {
        float a = b1[j];
        const float* wr = W1 + j * 100;
        #pragma unroll 4
        for (int c = 0; c < 100; c++) a = fmaf(wr[c], xr[c], a);
        a = fmaxf(a, 0.f);
        y = fmaf(W2[j], a, y);
    }
    logits[bt] = y + mask[bt];
}

// alpha = softmax(logits, axis=t); one block per batch row
__global__ void k_alpha(const float* __restrict__ logits, float* __restrict__ alpha) {
    __shared__ float red[256];
    int b = blockIdx.x, tid = threadIdx.x;
    const float* lp = logits + (size_t)b * TT;
    float lm = -1e30f;
    for (int t = tid; t < TT; t += 256) lm = fmaxf(lm, lp[t]);
    red[tid] = lm; __syncthreads();
    for (int s = 128; s > 0; s >>= 1) {
        if (tid < s) red[tid] = fmaxf(red[tid], red[tid + s]);
        __syncthreads();
    }
    float mx = red[0]; __syncthreads();
    float ls = 0.f;
    for (int t = tid; t < TT; t += 256) ls += __expf(lp[t] - mx);
    red[tid] = ls; __syncthreads();
    for (int s = 128; s > 0; s >>= 1) {
        if (tid < s) red[tid] += red[tid + s];
        __syncthreads();
    }
    float inv = 1.f / red[0];
    for (int t = tid; t < TT; t += 256)
        alpha[(size_t)b * TT + t] = __expf(lp[t] - mx) * inv;
}

// ---------------------------------------------------------------------------
extern "C" void kernel_launch(void* const* d_in, const int* in_sizes, int n_in,
                              void* d_out, int out_size, void* d_ws, size_t ws_size,
                              hipStream_t stream) {
    (void)in_sizes; (void)n_in; (void)out_size; (void)ws_size;
    const int*   x    = (const int*)  d_in[0];
    const float* mask = (const float*)d_in[1];
    const float* emb  = (const float*)d_in[2];
    const float* Wq   = (const float*)d_in[3];
    const float* Wk   = (const float*)d_in[4];
    const float* Wv   = (const float*)d_in[5];
    const float* Wu   = (const float*)d_in[6];
    const float* bu   = (const float*)d_in[7];
    const float* W1   = (const float*)d_in[8];
    const float* b1   = (const float*)d_in[9];
    const float* W2   = (const float*)d_in[10];
    const float* b2   = (const float*)d_in[11];

    char* ws = (char*)d_ws;                        // ~84.3 MB total
    half_t* h16    = (half_t*)(ws + 0);            // 16 MB  [BT][128] f16
    half_t* qb     = (half_t*)(ws + 16777216);     // 16 MB  [64][2048][64] f16
    half_t* kb     = (half_t*)(ws + 33554432);     // 16 MB  [64][2048][64] f16
    half_t* vtb    = (half_t*)(ws + 50331648);     // 16 MB  [64][64][2048] f16 (V^T)
    half_t* ctxh   = (half_t*)(ws + 67108864);     // 16 MB  [BT][128] f16
    half_t* wpack  = (half_t*)(ws + 83886080);     // 112 KB [4][112][128] f16
    float*  logits = (float*) (ws + 84000768);     // 256 KB [BT] f32

    float* alpha = (float*)d_out;                  // [32][2048]
    float* x1    = (float*)d_out + BT;             // [32][2048][100]

    // zero q/k/v^T/ctx pads (contiguous 64 MB span starting at qb)
    k_zero<<<16384, 256, 0, stream>>>((uint4*)qb, 4194304);
    k_prep_w<<<(4 * NPAD * DPAD + 255) / 256, 256, 0, stream>>>(Wq, Wk, Wv, Wu, wpack);
    k_build_h<<<BT * DPAD / 256, 256, 0, stream>>>(x, emb, h16);
    k_qkv<<<dim3(7168, 3), 128, 0, stream>>>(h16, wpack, qb, kb, vtb);
    k_attn<<<2048, 128, 0, stream>>>(qb, kb, vtb, mask, ctxh);
    k_wu<<<7168, 128, 0, stream>>>(ctxh, wpack, bu, x, emb, x1);
    k_mlp<<<BT / 256, 256, 0, stream>>>(x1, W1, b1, W2, b2, mask, logits);
    k_alpha<<<BB, 256, 0, stream>>>(logits, alpha);
}
